// Covariance_Metric_23656679866493
// MI455X (gfx1250) — matrix operand order repaired and checked by reading, not compiled, not run on hardware
//
#include <hip/hip_runtime.h>

#define WAY 5
#define SHOT 5
#define CH 64
#define HW 441
#define HWP 448
#define NQ 512
#define NLOC (SHOT*HW)      // 2205
#define QSTRIDE 72          // padded LDS row stride in ushorts (144B: 16B aligned, bank-friendly)

typedef __attribute__((ext_vector_type(16))) __bf16        bf16x16;
typedef __attribute__((ext_vector_type(8)))  float         f32x8;
typedef __attribute__((ext_vector_type(4)))  unsigned int  u32x4;

union BfVec { u32x4 u[2]; bf16x16 v; };

__device__ __forceinline__ unsigned short f2bf(float f) {
  unsigned int u = __float_as_uint(f);
  u += 0x7FFFu + ((u >> 16) & 1u);      // round-to-nearest-even
  return (unsigned short)(u >> 16);
}
__device__ __forceinline__ float bf2f(unsigned short h) {
  return __uint_as_float(((unsigned int)h) << 16);
}

// gfx1250 async global->LDS 16B copy (ASYNCcnt). ldsOff = byte offset from LDS_BASE.
__device__ __forceinline__ void async_ldsb128(unsigned int ldsOff, const void* gsrc) {
  asm volatile("global_load_async_to_lds_b128 %0, %1, off"
               :: "v"(ldsOff), "v"(gsrc) : "memory");
}
__device__ __forceinline__ void wait_asynccnt0() {
  asm volatile("s_wait_asynccnt 0x0" ::: "memory");
}

// ---- Stage 1a: per-class per-channel mean of support descriptors ----
__global__ void support_mean_kernel(const float* __restrict__ x2, float* __restrict__ means) {
  const int w = blockIdx.x;        // 0..4
  const int c = threadIdx.x;       // 0..63
  float sum = 0.f;
  for (int shot = 0; shot < SHOT; ++shot) {
    const float* p = x2 + ((size_t)(w*SHOT + shot)*CH + c)*HW;
    for (int s = 0; s < HW; ++s) sum += p[s];
  }
  means[w*CH + c] = sum * (1.0f/(float)NLOC);
}

// ---- Stage 1b: covariance (5 x 64 x 64), written as bf16 for WMMA A-operand ----
__global__ void support_cov_kernel(const float* __restrict__ x2,
                                   const float* __restrict__ means,
                                   unsigned short* __restrict__ cov_bf16) {
  __shared__ float fc[NLOC];
  const int w = blockIdx.x / CH;
  const int c = blockIdx.x % CH;
  const int tid = threadIdx.x;     // 64 threads
  const float mc = means[w*CH + c];
  for (int i = tid; i < NLOC; i += CH) {
    int shot = i / HW, s = i - shot*HW;
    fc[i] = x2[((size_t)(w*SHOT + shot)*CH + c)*HW + s] - mc;
  }
  __syncthreads();
  const int d = tid;
  const float md = means[w*CH + d];
  float sum = 0.f;
  for (int shot = 0; shot < SHOT; ++shot) {
    const float* p = x2 + ((size_t)(w*SHOT + shot)*CH + d)*HW;
    const float* q = fc + shot*HW;
    for (int s = 0; s < HW; ++s) sum += q[s] * (p[s] - md);
  }
  cov_bf16[((size_t)w*CH + c)*CH + d] = f2bf(sum * (1.0f/(float)(NLOC - 1)));
}

// ---- Stage 2: fused center + v^T Cov v (WMMA) + leaky-relu conv head ----
__global__ __launch_bounds__(128) void covsim_kernel(
    const float* __restrict__ x1, const unsigned short* __restrict__ cov_bf16,
    const float* __restrict__ conv_w, const float* __restrict__ conv_b,
    float* __restrict__ out) {
  extern __shared__ unsigned char smem[];
  unsigned short* sQt  = (unsigned short*)smem;                    // [448][72] bf16, q^T centered
  unsigned short* sCov = (unsigned short*)(smem + 64512);          // [5][64][72] bf16
  float*          sCW  = (float*)(smem + 64512 + 46080);           // [448] conv_w padded
  float*          sRed = (float*)(smem + 64512 + 46080 + 1792);    // [4][5]

  const int tid  = threadIdx.x;
  const int wave = tid >> 5;
  const int lane = tid & 31;
  const int half = lane >> 4;        // 0: lanes 0-15, 1: lanes 16-31
  const int l16  = lane & 15;
  const int b    = blockIdx.x;

  // Prefetch this block's x1 slice (112896 B) so the centering pass hits cache;
  // overlaps with the async cov copy below. 128B-line granularity.
  {
    const float* xb = x1 + (size_t)b*CH*HW;
    for (int i = tid; i < (CH*HW*4)/128; i += 128)
      __builtin_prefetch(xb + i*32, 0, 3);
  }

  // Async global->LDS copy of cov (5*64 rows x 128B) into padded-stride LDS.
  // 2560 x 16B chunks, 20 per thread, runs on ASYNCcnt under the centering pass.
  for (int k = tid; k < (WAY*CH*CH)/8; k += 128) {
    int row  = k >> 3;               // w*64 + c
    int col8 = k & 7;
    unsigned int ldsOff = 64512u + (unsigned int)(row*QSTRIDE + col8*8)*2u;
    async_ldsb128(ldsOff, cov_bf16 + row*CH + col8*8);
  }

  // stage conv_w with zero padding to 448 (tiny, regular path)
  for (int i = tid; i < HWP; i += 128) sCW[i] = (i < HW) ? conv_w[i] : 0.f;

  // center q over spatial dim, store transposed bf16: sQt[s][c]
  for (int c = wave*16; c < wave*16 + 16; ++c) {
    const float* p = x1 + ((size_t)b*CH + c)*HW;
    float vals[14];
    float sum = 0.f;
#pragma unroll
    for (int i = 0; i < 14; ++i) {
      int idx = i*32 + lane;
      float v = (idx < HW) ? p[idx] : 0.f;
      vals[i] = v;
      sum += v;
    }
#pragma unroll
    for (int off = 16; off >= 1; off >>= 1) sum += __shfl_xor(sum, off, 32);
    const float mean = sum * (1.0f/(float)HW);
#pragma unroll
    for (int i = 0; i < 14; ++i) {
      int idx = i*32 + lane;       // 0..447, full coverage incl. zero padding
      sQt[idx*QSTRIDE + c] = (idx < HW) ? f2bf(vals[i] - mean) : (unsigned short)0;
    }
  }

  wait_asynccnt0();                 // cov tiles landed in LDS
  __syncthreads();

  float a0 = 0.f, a1 = 0.f, a2 = 0.f, a3 = 0.f, a4 = 0.f;

  // 140 wave-jobs: (class w, 16-column chunk); 4 waves -> 35 each, no divergence around WMMA
  for (int job = wave; job < WAY*28; job += 4) {
    const int w     = job / 28;
    const int chunk = job - w*28;
    const int s     = chunk*16 + l16;                 // this lane's column (spatial index)
    const unsigned short* qrow = sQt + s*QSTRIDE;
    const unsigned short* covb = sCov + w*CH*QSTRIDE;

    // B operands (Q^T tile, 32x16): lanes 0-15 hold K 0-15, lanes 16-31 hold K 16-31
    BfVec B0, B1;
    B0.u[0] = *(const u32x4*)(qrow + half*16 + 0);
    B0.u[1] = *(const u32x4*)(qrow + half*16 + 8);
    B1.u[0] = *(const u32x4*)(qrow + 32 + half*16 + 0);
    B1.u[1] = *(const u32x4*)(qrow + 32 + half*16 + 8);

    float partial = 0.f;
#pragma unroll
    for (int rt = 0; rt < 4; ++rt) {
      // A operand (Cov tile 16x32): lane holds row M = rt*16 + l16,
      // K 0-15 (lanes 0-15) / K 16-31 (lanes 16-31)
      const unsigned short* arow = covb + (rt*16 + l16)*QSTRIDE;
      BfVec A0, A1;
      A0.u[0] = *(const u32x4*)(arow + half*16 + 0);
      A0.u[1] = *(const u32x4*)(arow + half*16 + 8);
      A1.u[0] = *(const u32x4*)(arow + 32 + half*16 + 0);
      A1.u[1] = *(const u32x4*)(arow + 32 + half*16 + 8);

      f32x8 acc = {0.f,0.f,0.f,0.f,0.f,0.f,0.f,0.f};
      acc = __builtin_amdgcn_wmma_f32_16x16x32_bf16(false, A0.v, false, B0.v,
                                                    (short)0, acc, false, false);
      acc = __builtin_amdgcn_wmma_f32_16x16x32_bf16(false, A1.v, false, B1.v,
                                                    (short)0, acc, false, false);

      // epilogue: sim partial = sum_r q[M(r), s] * T[M(r), s];
      // D VGPR r holds M = rt*16 + r + half*8, N = l16
      u32x4 qv = *(const u32x4*)(qrow + rt*16 + half*8);
#pragma unroll
      for (int j = 0; j < 4; ++j) {
        unsigned int pk = qv[j];
        partial += bf2f((unsigned short)(pk & 0xFFFFu)) * acc[2*j + 0];
        partial += bf2f((unsigned short)(pk >> 16))     * acc[2*j + 1];
      }
    }
    // combine channel halves held by lanes L and L^16 (same column s)
    partial += __shfl_xor(partial, 16, 32);

    float contrib = 0.f;
    if (lane < 16) {
      float v = (partial > 0.f) ? partial : 0.2f * partial;   // LeakyReLU(0.2)
      contrib = v * sCW[s];                                   // zero-padded past 441
    }
    if      (w == 0) a0 += contrib;
    else if (w == 1) a1 += contrib;
    else if (w == 2) a2 += contrib;
    else if (w == 3) a3 += contrib;
    else             a4 += contrib;
  }

  // wave-level reduction (lanes >=16 carry zeros)
#pragma unroll
  for (int off = 16; off >= 1; off >>= 1) {
    a0 += __shfl_xor(a0, off, 32);
    a1 += __shfl_xor(a1, off, 32);
    a2 += __shfl_xor(a2, off, 32);
    a3 += __shfl_xor(a3, off, 32);
    a4 += __shfl_xor(a4, off, 32);
  }
  if (lane == 0) {
    sRed[wave*5+0] = a0; sRed[wave*5+1] = a1; sRed[wave*5+2] = a2;
    sRed[wave*5+3] = a3; sRed[wave*5+4] = a4;
  }
  __syncthreads();
  if (tid < WAY) {
    float t = sRed[tid] + sRed[5+tid] + sRed[10+tid] + sRed[15+tid] + conv_b[0];
    out[(size_t)b*WAY + tid] = t;
  }
}

extern "C" void kernel_launch(void* const* d_in, const int* in_sizes, int n_in,
                              void* d_out, int out_size, void* d_ws, size_t ws_size,
                              hipStream_t stream) {
  (void)in_sizes; (void)n_in; (void)out_size; (void)ws_size;
  const float* x1     = (const float*)d_in[0];
  const float* x2     = (const float*)d_in[1];
  const float* conv_w = (const float*)d_in[2];
  const float* conv_b = (const float*)d_in[3];
  float* out = (float*)d_out;

  float*          means = (float*)d_ws;                                          // 5*64 f32
  unsigned short* cov   = (unsigned short*)((char*)d_ws + WAY*CH*sizeof(float)); // 5*64*64 bf16

  support_mean_kernel<<<WAY, CH, 0, stream>>>(x2, means);
  support_cov_kernel<<<WAY*CH, CH, 0, stream>>>(x2, means, cov);

  const size_t lds_bytes = 64512 + 46080 + 1792 + 80;   // ~112.5 KB of 320 KB/WGP
  covsim_kernel<<<NQ, 128, lds_bytes, stream>>>(x1, cov, conv_w, conv_b, out);
}
